// Step_29575144800975
// MI455X (gfx1250) — matrix operand order repaired and checked by reading, not compiled, not run on hardware
//
#include <hip/hip_runtime.h>
#include <hip/hip_bf16.h>
#include <math.h>

typedef __attribute__((ext_vector_type(16))) __bf16 v16bf;
typedef __attribute__((ext_vector_type(8)))  float  v8f;

#define BATCH   4
#define SEQ     8192
#define DIM     1024
#define ROWS    (BATCH * SEQ)            // 32768
#define NTOT    ((size_t)ROWS * DIM)     // 33554432 elements per output tensor
#define NCHUNK  64
#define CHLEN   (SEQ / NCHUNK)           // 128
#define NCHAN   (BATCH * DIM)            // 4096

// ---------- helpers ----------
static __device__ __forceinline__ unsigned short f2bf(float f) {
  unsigned u = __builtin_bit_cast(unsigned, f);
  unsigned r = ((u >> 16) & 1u) + 0x7fffu;   // round-to-nearest-even
  u += r;
  return (unsigned short)(u >> 16);
}

static __device__ __forceinline__ float sigmoidf(float z) {
  return 1.0f / (1.0f + __expf(-z));
}

// ---------- 1) pack W (1024x1024 f32 row-major KxN) into WMMA-B fragment order (bf16) ----------
// Contiguous-K layout used here (B 32x16, wave32): lane<16 -> col = nt*16+lane, dword v holds K pair
// (kt*32 + 2v, +1); lane>=16 -> col = nt*16+lane-16, K pair (kt*32+16+2v, +1).
// Stored per fragment: 32 lanes x 8 dwords contiguous -> each lane loads 2x uint4.
__global__ void pack_w_kernel(const float* __restrict__ W, unsigned* __restrict__ pack) {
  int dw   = blockIdx.x * blockDim.x + threadIdx.x;   // 0 .. 524287
  int v    = dw & 7;
  int lane = (dw >> 3) & 31;
  int frag = dw >> 8;                                  // kt*64 + nt
  int nt   = frag & 63;
  int kt   = frag >> 6;
  int col  = nt * 16 + (lane & 15);
  int k0   = kt * 32 + ((lane >> 4) << 4) + 2 * v;
  unsigned lo = f2bf(W[(size_t)k0 * DIM + col]);
  unsigned hi = f2bf(W[(size_t)(k0 + 1) * DIM + col]);
  pack[dw] = lo | (hi << 16);
}

// ---------- 2) gate_input = rmsnorm(c)*sc + rmsnorm(prev)*sp*g_prev (bf16 out) ----------
__global__ void gate_kernel(const float* __restrict__ c, const float* __restrict__ prev,
                            const float* __restrict__ sc, const float* __restrict__ sp,
                            const float* __restrict__ gp, const int* __restrict__ step,
                            unsigned short* __restrict__ gate) {
  __shared__ float red1[256];
  __shared__ float red2[256];
  int row = blockIdx.x;
  int t   = threadIdx.x;
  size_t base = (size_t)row * DIM;
  float cv[4], pv[4];
  float s1 = 0.f, s2 = 0.f;
#pragma unroll
  for (int i = 0; i < 4; ++i) {
    int d = t + 256 * i;
    cv[i] = c[base + d];
    pv[i] = prev[base + d];
    s1 += cv[i] * cv[i];
    s2 += pv[i] * pv[i];
  }
  red1[t] = s1; red2[t] = s2;
  __syncthreads();
  for (int off = 128; off > 0; off >>= 1) {
    if (t < off) { red1[t] += red1[t + off]; red2[t] += red2[t + off]; }
    __syncthreads();
  }
  float rc = rsqrtf(red1[0] * (1.0f / DIM) + 1e-6f);
  float rp = rsqrtf(red2[0] * (1.0f / DIM) + 1e-6f);
  float use_prev = (step[0] != 0) ? 1.0f : 0.0f;
#pragma unroll
  for (int i = 0; i < 4; ++i) {
    int d = t + 256 * i;
    float g = cv[i] * rc * sc[d] + use_prev * (pv[i] * rp * sp[d] * gp[d]);
    gate[base + d] = f2bf(g);
  }
}

// ---------- 3) dual GEMM (alpha & beta share the A operand) + fused nonlinearity ----------
// Block = 128 threads (4 waves). Block tile: M=16 rows, N=256 cols (wave w -> 4 N-tiles of 16).
// K loop = 32 iters of v_wmma_f32_16x16x32_bf16 per accumulator, A staged via LDS.
__global__ __launch_bounds__(128) void gemm_gate_kernel(
    const unsigned short* __restrict__ gate,   // ROWS x DIM bf16
    const unsigned* __restrict__ packA,        // W_alpha, fragment-packed bf16
    const unsigned* __restrict__ packB,        // W_beta,  fragment-packed bf16
    const float* __restrict__ b_alpha, const float* __restrict__ b_beta,
    const float* __restrict__ vin,
    float* __restrict__ alpha_out,             // staged in d_out's out_new region
    float* __restrict__ x_out)                 // staged in d_out's fetched region
{
  __shared__ uint2 ldsA[128];                  // 16 rows x 32 bf16 (64B row stride)
  const int tid  = threadIdx.x;
  const int wave = tid >> 5;
  const int lane = tid & 31;
  const int mt   = blockIdx.y;
  const int nt0  = blockIdx.x * 16 + wave * 4; // first of 4 contiguous N-tiles
  const int r    = lane & 15;
  const int hl   = lane >> 4;

  v8f zero = {0.f, 0.f, 0.f, 0.f, 0.f, 0.f, 0.f, 0.f};
  v8f accA[4], accB[4];
#pragma unroll
  for (int j = 0; j < 4; ++j) { accA[j] = zero; accB[j] = zero; }

  const unsigned* gatedw = (const unsigned*)gate;      // bf16 pairs, row stride 512 dwords

  for (int kt = 0; kt < 32; ++kt) {
    __syncthreads();
    {
      int i    = tid * 2;          // dword index in 16x32 tile (256 dwords)
      int row  = i >> 4;
      int cold = i & 15;
      ldsA[tid] = *(const uint2*)(gatedw + (size_t)(mt * 16 + row) * 512 + kt * 16 + cold);
    }
    __syncthreads();

    // A fragment (contiguous K, matches packed B): lane<16 K{0..15}, lane>=16 K{16..31}
    union { uint4 q[2]; v16bf v; } af;
    const char* lb = (const char*)ldsA;
    af.q[0] = *(const uint4*)(lb + r * 64 + hl * 32);
    af.q[1] = *(const uint4*)(lb + r * 64 + hl * 32 + 16);

#pragma unroll
    for (int j = 0; j < 4; ++j) {
      size_t fb = ((size_t)(kt * 64 + (nt0 + j)) * 32 + lane) * 8;
      union { uint4 q[2]; v16bf v; } bfa;
      const uint4* pa = (const uint4*)(packA + fb);
      bfa.q[0] = pa[0]; bfa.q[1] = pa[1];
      accA[j] = __builtin_amdgcn_wmma_f32_16x16x32_bf16(
          false, af.v, false, bfa.v, (short)0, accA[j], false, false);

      union { uint4 q[2]; v16bf v; } bfb;
      const uint4* pb = (const uint4*)(packB + fb);
      bfb.q[0] = pb[0]; bfb.q[1] = pb[1];
      accB[j] = __builtin_amdgcn_wmma_f32_16x16x32_bf16(
          false, af.v, false, bfb.v, (short)0, accB[j], false, false);
    }
  }

  // Epilogue: alpha = sigmoid(zA+bA); x = v * silu(zB+bB) * sqrt(clip(1-alpha^2, 1e-6))
#pragma unroll
  for (int j = 0; j < 4; ++j) {
    int n   = (nt0 + j) * 16 + r;
    float ba = b_alpha[n];
    float bb = b_beta[n];
#pragma unroll
    for (int vv = 0; vv < 8; ++vv) {
      int m = mt * 16 + hl * 8 + vv;         // C/D layout: lane<16 M=v, lane>=16 M=v+8
      float za = accA[j][vv] + ba;
      float zb = accB[j][vv] + bb;
      float al = sigmoidf(za);
      float sl = zb * sigmoidf(zb);
      float ws = sqrtf(fmaxf(1.0f - al * al, 1e-6f));
      size_t idx = (size_t)m * DIM + n;
      alpha_out[idx] = al;
      x_out[idx]     = vin[idx] * sl * ws;
    }
  }
}

// ---------- 4) local scan per (channel, chunk): h_t = a_t*h_{t-1} + x_t ----------
__global__ void scan_local_kernel(const float* __restrict__ alpha, float* xl,
                                  float* __restrict__ aggP, float* __restrict__ aggH) {
  int d  = blockIdx.x * blockDim.x + threadIdx.x;
  int b  = blockIdx.y;
  int ch = blockIdx.z;
  size_t base = ((size_t)(b * SEQ + ch * CHLEN)) * DIM + d;
  float h = 0.f, p = 1.f;
  for (int t = 0; t < CHLEN; ++t) {
    size_t idx = base + (size_t)t * DIM;
    float a = alpha[idx];
    float x = xl[idx];
    h = fmaf(a, h, x);
    p *= a;
    xl[idx] = h;                     // overwrite x with local-scan value
  }
  int chan = b * DIM + d;
  aggP[(size_t)ch * NCHAN + chan] = p;
  aggH[(size_t)ch * NCHAN + chan] = h;
}

// ---------- 5) sequential scan over the 64 chunk aggregates per channel ----------
__global__ void scan_chunk_kernel(const float* __restrict__ aggP, const float* __restrict__ aggH,
                                  float* __restrict__ carry) {
  int chan = blockIdx.x * blockDim.x + threadIdx.x;   // 0..4095
  float C = 0.f;
  for (int ch = 0; ch < NCHUNK; ++ch) {
    carry[(size_t)ch * NCHAN + chan] = C;
    C = fmaf(aggP[(size_t)ch * NCHAN + chan], C, aggH[(size_t)ch * NCHAN + chan]);
  }
}

// ---------- 6) apply carries + fused output epilogue ----------
// alpha_out doubles as the out_new output buffer; fl doubles as the fetched output buffer.
// Reads at idx strictly precede writes at idx within each thread.
__global__ void scan_apply_kernel(float* alpha_out, float* fl,
                                  const float* __restrict__ carry,
                                  const float* __restrict__ out_in,
                                  const float* __restrict__ vin,
                                  const float* __restrict__ cin,
                                  const float* __restrict__ g_out,
                                  const float* __restrict__ g_v,
                                  float* __restrict__ v_new,
                                  float* __restrict__ c_out) {
  int d  = blockIdx.x * blockDim.x + threadIdx.x;
  int b  = blockIdx.y;
  int ch = blockIdx.z;
  int chan = b * DIM + d;
  float C  = carry[(size_t)ch * NCHAN + chan];
  float go = g_out[d], gv = g_v[d];
  size_t base = ((size_t)(b * SEQ + ch * CHLEN)) * DIM + d;
  float p = 1.f;
  for (int t = 0; t < CHLEN; ++t) {
    size_t idx = base + (size_t)t * DIM;
    float a = alpha_out[idx];
    float l = fl[idx];
    p *= a;
    float f = fmaf(p, C, l);         // true h_t = l_t + (prod alpha) * carry
    fl[idx]        = f;              // fetched
    alpha_out[idx] = out_in[idx] + f * go;   // out_new (overwrites staged alpha)
    v_new[idx]     = vin[idx] + f * gv;
    c_out[idx]     = cin[idx];
  }
}

extern "C" void kernel_launch(void* const* d_in, const int* in_sizes, int n_in,
                              void* d_out, int out_size, void* d_ws, size_t ws_size,
                              hipStream_t stream) {
  const float* v       = (const float*)d_in[0];
  const float* out     = (const float*)d_in[1];
  const float* c       = (const float*)d_in[2];
  const float* prev    = (const float*)d_in[3];
  const float* rms_c   = (const float*)d_in[4];
  const float* rms_p   = (const float*)d_in[5];
  const float* g_prev  = (const float*)d_in[6];
  const float* W_alpha = (const float*)d_in[7];
  const float* b_alpha = (const float*)d_in[8];
  const float* W_beta  = (const float*)d_in[9];
  const float* b_beta  = (const float*)d_in[10];
  const float* g_out   = (const float*)d_in[11];
  const float* g_v     = (const float*)d_in[12];
  const int*   step    = (const int*)d_in[13];

  // Workspace layout (bytes): gate bf16 64MB | packA 2MB | packB 2MB | aggP 1MB | aggH 1MB | carry 1MB
  char* ws = (char*)d_ws;
  unsigned short* gate  = (unsigned short*)ws;
  unsigned*       packA = (unsigned*)(ws + 67108864);
  unsigned*       packB = (unsigned*)(ws + 69206016);
  float*          aggP  = (float*)(ws + 71303168);
  float*          aggH  = (float*)(ws + 72351744);
  float*          carry = (float*)(ws + 73400320);

  float* o       = (float*)d_out;
  float* v_new   = o;                 // output 0
  float* out_new = o + NTOT;          // output 1 (doubles as alpha staging)
  float* c_out   = o + 2 * NTOT;      // output 2
  float* fetched = o + 3 * NTOT;      // output 3 (doubles as x / local-scan staging)

  pack_w_kernel<<<2048, 256, 0, stream>>>(W_alpha, packA);
  pack_w_kernel<<<2048, 256, 0, stream>>>(W_beta,  packB);
  gate_kernel<<<ROWS, 256, 0, stream>>>(c, prev, rms_c, rms_p, g_prev, step, gate);
  gemm_gate_kernel<<<dim3(4, 2048), 128, 0, stream>>>(gate, packA, packB, b_alpha, b_beta,
                                                      v, out_new, fetched);
  scan_local_kernel<<<dim3(4, BATCH, NCHUNK), 256, 0, stream>>>(out_new, fetched, aggP, aggH);
  scan_chunk_kernel<<<16, 256, 0, stream>>>(aggP, aggH, carry);
  scan_apply_kernel<<<dim3(4, BATCH, NCHUNK), 256, 0, stream>>>(out_new, fetched, carry,
                                                                out, v, c, g_out, g_v,
                                                                v_new, c_out);
}
